// EnhancedStrategySuperposition_18923625906344
// MI455X (gfx1250) — matrix operand order repaired and checked by reading, not compiled, not run on hardware
//
#include <hip/hip_runtime.h>
#include <hip/hip_bf16.h>
#include <stdint.h>

typedef _Float16 h16;
typedef uint32_t u32;
typedef __attribute__((ext_vector_type(4)))  _Float16 v4h;
typedef __attribute__((ext_vector_type(8)))  _Float16 v8h;
typedef __attribute__((ext_vector_type(16))) _Float16 v16h;
typedef __attribute__((ext_vector_type(8)))  float    v8f;
typedef __attribute__((ext_vector_type(4)))  u32      u32x4;
typedef __attribute__((ext_vector_type(8)))  int      i32x8;
typedef __attribute__((ext_vector_type(4)))  int      i32x4;

#define B_TOTAL 32768
#define D_STATE 512
#define NEXP    16
#define H1      128
#define H2      64
#define ADIM    64

#define THREADS 128
#define WAVES   4
#define ROWS_PER_BLOCK 64
#define NBLOCKS (B_TOTAL / ROWS_PER_BLOCK)   // 512
#define NSLOTS  (NEXP * 2)                   // 2 half-panels of W1 per expert

// ---- LDS layout (units: halves). Row strides padded (+8) to rotate banks. ----
#define SA_STRIDE  520
#define SW1_STRIDE 520
#define SW2_STRIDE 136
#define SW3_STRIDE 72
#define SH_STRIDE  136
#define SA_BASE  0
#define W1HALF   (64 * SW1_STRIDE)                       // 33280 halves per half-panel
#define W1B_BASE (SA_BASE + ROWS_PER_BLOCK * SA_STRIDE)  // 33280 (buf0 / buf1 ping-pong)
#define W2BUF    (H2 * SW2_STRIDE)                       // 8704
#define W2B_BASE (W1B_BASE + 2 * W1HALF)                 // 99840
#define W3BUF    (ADIM * SW3_STRIDE)                     // 4608
#define W3B_BASE (W2B_BASE + 2 * W2BUF)                  // 117248
#define SH_BASE  (W3B_BASE + 2 * W3BUF)                  // 126464
#define SMEM_HALVES (SH_BASE + WAVES * 16 * SH_STRIDE)   // 135168
#define SMEM_BYTES  (SMEM_HALVES * 2)                    // 270336 <= 320KB
static_assert(SMEM_BYTES <= 320 * 1024, "LDS budget");

// Workspace layout (bytes)
#define WS_W1T   0
#define WS_W2T   (WS_W1T + NEXP * H1 * D_STATE * 2)   // 2097152
#define WS_W3T   (WS_W2T + NEXP * H2 * H1 * 2)        // +262144
#define WS_GWT   (WS_W3T + NEXP * ADIM * H2 * 2)      // +131072
#define WS_STATE (WS_GWT + NEXP * D_STATE * 2)        // +16384 (4KB aligned)
#define WS_END   (WS_STATE + B_TOTAL * D_STATE * 2)   // ~36MB total

// ---------------------------------------------------------------------------
// Kernel 1: fp32 weights -> f16 transposed "B" layout; state -> f16.
// ---------------------------------------------------------------------------
__global__ void pack_inputs_kernel(const float* __restrict__ state,
                                   const float* __restrict__ W1,
                                   const float* __restrict__ W2,
                                   const float* __restrict__ W3,
                                   const float* __restrict__ gW,
                                   h16* __restrict__ w1t, h16* __restrict__ w2t,
                                   h16* __restrict__ w3t, h16* __restrict__ gwt,
                                   h16* __restrict__ stf) {
  const int tid  = blockIdx.x * blockDim.x + threadIdx.x;
  const int nthr = gridDim.x * blockDim.x;
  for (int i = tid; i < B_TOTAL * D_STATE / 4; i += nthr) {
    const float4 v = reinterpret_cast<const float4*>(state)[i];
    v4h h = { (h16)v.x, (h16)v.y, (h16)v.z, (h16)v.w };
    reinterpret_cast<v4h*>(stf)[i] = h;
  }
  for (int i = tid; i < NEXP * H1 * D_STATE; i += nthr) {
    int k = i / (H1 * D_STATE), r = i % (H1 * D_STATE);
    int h = r / D_STATE, d = r % D_STATE;
    w1t[i] = (h16)W1[(k * D_STATE + d) * H1 + h];
  }
  for (int i = tid; i < NEXP * H2 * H1; i += nthr) {
    int k = i / (H2 * H1), r = i % (H2 * H1);
    int m = r / H1, h = r % H1;
    w2t[i] = (h16)W2[(k * H1 + h) * H2 + m];
  }
  for (int i = tid; i < NEXP * ADIM * H2; i += nthr) {
    int k = i / (ADIM * H2), r = i % (ADIM * H2);
    int a = r / H2, m = r % H2;
    w3t[i] = (h16)W3[(k * H2 + m) * ADIM + a];
  }
  for (int i = tid; i < NEXP * D_STATE; i += nthr) {
    int j = i / D_STATE, d = i % D_STATE;
    gwt[i] = (h16)gW[d * NEXP + j];
  }
}

// ---------------------------------------------------------------------------
// TDM: one descriptor moves a 2D panel Global->LDS, inserting the +8-half
// row padding via pad_interval/pad_amount. Tracked by TENSORcnt (in order
// per wave, so wait(1) after issuing panel t+1 proves panel t has landed).
// ---------------------------------------------------------------------------
#if __has_builtin(__builtin_amdgcn_tensor_load_to_lds)
#define HAVE_TDM 1
__device__ __forceinline__ void tdm_load_2d(h16* lds_ptr, const h16* g_ptr,
                                            u32 d0, u32 rows, u32 g_stride,
                                            u32 pad_interval, u32 pad_amount) {
  const uint64_t ga = (uint64_t)(uintptr_t)g_ptr;
  u32x4 g0;
  g0[0] = 1u;                                    // count=1, user mode, no gather
  g0[1] = (u32)(uintptr_t)lds_ptr;               // LDS byte address
  g0[2] = (u32)(ga & 0xFFFFFFFFu);               // global_addr[31:0]
  g0[3] = (u32)((ga >> 32) & 0x01FFFFFFu) | (2u << 30);  // addr[56:32] | type=2
  i32x8 g1;
  g1[0] = (int)((1u << 16) |                     // data_size = 1 (2 bytes)
                (1u << 20) |                     // pad_enable
                (pad_interval << 22) | (pad_amount << 25));
  g1[1] = (int)((d0 & 0xFFFFu) << 16);           // tensor_dim0 lo (barrier addr=0)
  g1[2] = (int)((d0 >> 16) | ((rows & 0xFFFFu) << 16));   // dim0 hi | tensor_dim1 lo
  g1[3] = (int)((rows >> 16) | ((d0 & 0xFFFFu) << 16));   // dim1 hi | tile_dim0
  g1[4] = (int)(rows & 0xFFFFu);                 // tile_dim1 (tile_dim2 = 0)
  g1[5] = (int)g_stride;                         // tensor_dim0_stride[31:0]
  g1[6] = 0;
  g1[7] = 0;
  i32x4 g2; g2[0] = 1; g2[1] = 1; g2[2] = 0; g2[3] = 0;  // tensor_dim2/3=1, tile_dim3=0
  i32x4 g3; g3[0] = 0; g3[1] = (int)(1u << 16); g3[2] = 0; g3[3] = 0;  // tensor_dim4=1
#if __clang_major__ >= 23
  i32x8 z8 = {0, 0, 0, 0, 0, 0, 0, 0};
  __builtin_amdgcn_tensor_load_to_lds(g0, g1, g2, g3, z8, 0);
#else
  __builtin_amdgcn_tensor_load_to_lds(g0, g1, g2, g3, 0);
#endif
}
#else
#define HAVE_TDM 0
#warning "CDNA5 TDM builtin __builtin_amdgcn_tensor_load_to_lds NOT available; using manual LDS staging fallback"
#endif

// ---------------------------------------------------------------------------
// WMMA fragment helpers (wave32 16x16x32 f16 layouts, CDNA5 ISA §7.12.2)
// ---------------------------------------------------------------------------
__device__ __forceinline__ v8h ld8(const h16* p) { return *reinterpret_cast<const v8h*>(p); }
__device__ __forceinline__ void st8(h16* p, v8h v) { *reinterpret_cast<v8h*>(p) = v; }
__device__ __forceinline__ v16h cat16(v8h a, v8h b) {
  return __builtin_shufflevector(a, b, 0,1,2,3,4,5,6,7,8,9,10,11,12,13,14,15);
}
__device__ __forceinline__ v16h ldA(const h16* buf, int row, int stride, int kb, int sel) {
  const h16* p = buf + row * stride + kb + sel * 8;
  return cat16(ld8(p), ld8(p + 16));
}
__device__ __forceinline__ v16h ldB(const h16* buf, int col, int stride, int kb, int sel) {
  const h16* p = buf + col * stride + kb + sel * 8;
  return cat16(ld8(p), ld8(p + 16));
}
#define WMMA_F16(a, b, c) \
  __builtin_amdgcn_wmma_f32_16x16x32_f16(false, (a), false, (b), (short)0, (c), false, false)

__device__ __forceinline__ float selu_f(float x) {
  const float kScale = 1.0507009873554805f;
  const float kAlpha = 1.6732632423543772f;
  float neg = kAlpha * (__expf(x) - 1.0f);
  return kScale * (x > 0.0f ? x : neg);
}

// ---------------------------------------------------------------------------
// Kernel 2: fused MoE, 4 waves/block, wave-per-16-row tile.
// TDM double-buffered pipeline: slot t computes W1 half-panel t while the
// DMA for half-panel t+1 (and next expert's W2/W3) streams into LDS.
// ---------------------------------------------------------------------------
__global__ void __launch_bounds__(THREADS, 1)
moe_fused_kernel(const h16* __restrict__ stf,
                 const float* __restrict__ b1,
                 const float* __restrict__ lng, const float* __restrict__ lnb,
                 const float* __restrict__ b2,  const float* __restrict__ b3,
                 const float* __restrict__ gate_b,
                 const h16* __restrict__ w1t, const h16* __restrict__ w2t,
                 const h16* __restrict__ w3t, const h16* __restrict__ gwt,
                 float* __restrict__ out) {
  extern __shared__ h16 smem[];
  h16* sA = smem + SA_BASE;

  const int tid  = threadIdx.x;
  const int wave = tid >> 5;
  const int lane = tid & 31;
  const int ln16 = lane & 15;
  const int sel  = lane >> 4;
  const int rowBase = blockIdx.x * ROWS_PER_BLOCK;
  h16* sH = smem + SH_BASE + wave * 16 * SH_STRIDE;   // wave-private 16x136

  // ---- prologue: stage state tile + first panels, overlapping gate GEMM ----
#if HAVE_TDM
  if (wave == 0) {
    tdm_load_2d(sA, stf + (size_t)rowBase * D_STATE, D_STATE, 32, D_STATE, 7, 3);
    tdm_load_2d(smem + W1B_BASE, w1t, D_STATE, 64, D_STATE, 7, 3);  // panel 0
    __builtin_amdgcn_s_wait_tensorcnt(1);   // state half done; panel 0 in flight
  } else if (wave == 1) {
    tdm_load_2d(sA + 32 * SA_STRIDE, stf + (size_t)(rowBase + 32) * D_STATE,
                D_STATE, 32, D_STATE, 7, 3);
    tdm_load_2d(smem + W2B_BASE, w2t, H1, H2, H1, 5, 3);            // W2(0)
    __builtin_amdgcn_s_wait_tensorcnt(1);   // state half done; W2(0) in flight
  } else if (wave == 2) {
    tdm_load_2d(smem + W3B_BASE, w3t, H2, ADIM, H2, 4, 3);          // W3(0)
  }
#else
  for (int c = tid; c < ROWS_PER_BLOCK * D_STATE / 8; c += THREADS) {
    int r = c >> 6, cc = (c & 63) << 3;
    st8(sA + r * SA_STRIDE + cc, ld8(stf + (size_t)(rowBase + r) * D_STATE + cc));
  }
#endif
  __syncthreads();

  // ---- gate logits: [16 x 512] x [512 x 16] via WMMA, B straight from L2 ----
  v8f g = {};
  for (int ks = 0; ks < 16; ++ks) {
    v16h a = ldA(sA, wave * 16 + ln16, SA_STRIDE, ks * 32, sel);
    v16h b = ldB(gwt, ln16, D_STATE, ks * 32, sel);
    g = WMMA_F16(a, b, g);
  }
  {
    float gb = gate_b[ln16];
    #pragma unroll
    for (int v = 0; v < 8; ++v) {
      float x = g[v] + gb;
      float mx = x;
      #pragma unroll
      for (int m = 8; m >= 1; m >>= 1) mx = fmaxf(mx, __shfl_xor(mx, m, 32));
      float e = __expf(x - mx);
      float ssum = e;
      #pragma unroll
      for (int m = 8; m >= 1; m >>= 1) ssum += __shfl_xor(ssum, m, 32);
      g[v] = e / ssum;   // softmax over 16 experts (N across the 16-lane group)
    }
  }

  v8f accOut[4] = {};  // gated output accumulator, 16x64 tile in D layout
  v8f h1[8];           // GEMM1 accumulators, persist across the 2 slots/expert

  for (int t = 0; t < NSLOTS; ++t) {
    const int k = t >> 1, half = t & 1;
    __syncthreads();  // everyone finished reading buffer (t-1); safe to overwrite
#if HAVE_TDM
    if (wave == 0) {
      if (t + 1 < NSLOTS) {   // issue next W1 half-panel, then prove panel t landed
        const int nk = (t + 1) >> 1, nh = (t + 1) & 1;
        tdm_load_2d(smem + W1B_BASE + nh * W1HALF,
                    w1t + (size_t)(nk * H1 + nh * 64) * D_STATE,
                    D_STATE, 64, D_STATE, 7, 3);
        __builtin_amdgcn_s_wait_tensorcnt(1);
      } else {
        __builtin_amdgcn_s_wait_tensorcnt(0);
      }
    } else if (wave == 1 && half == 0) {
      if (k + 1 < NEXP) {
        tdm_load_2d(smem + W2B_BASE + ((k + 1) & 1) * W2BUF,
                    w2t + (size_t)(k + 1) * H2 * H1, H1, H2, H1, 5, 3);
        __builtin_amdgcn_s_wait_tensorcnt(1);
      } else {
        __builtin_amdgcn_s_wait_tensorcnt(0);
      }
    } else if (wave == 2 && half == 0) {
      if (k + 1 < NEXP) {
        tdm_load_2d(smem + W3B_BASE + ((k + 1) & 1) * W3BUF,
                    w3t + (size_t)(k + 1) * ADIM * H2, H2, ADIM, H2, 4, 3);
        __builtin_amdgcn_s_wait_tensorcnt(1);
      } else {
        __builtin_amdgcn_s_wait_tensorcnt(0);
      }
    }
    if (half == 0 && k + 1 < NEXP) {  // warm L2 for the next expert's big panel
      __builtin_prefetch(w1t + (size_t)(k + 1) * H1 * D_STATE + tid * 512, 0, 1);
    }
#else
    {  // manual staging of this slot's half-panel (+ W2/W3 on even slots)
      const h16* src = w1t + (size_t)(k * H1 + half * 64) * D_STATE;
      h16* dst = smem + W1B_BASE + half * W1HALF;
      for (int c = tid; c < 64 * D_STATE / 8; c += THREADS) {
        int r = c >> 6, cc = (c & 63) << 3;
        st8(dst + r * SW1_STRIDE + cc, ld8(src + r * D_STATE + cc));
      }
      if (half == 0) {
        const h16* s2 = w2t + (size_t)k * H2 * H1;
        h16* d2 = smem + W2B_BASE + (k & 1) * W2BUF;
        for (int c = tid; c < H2 * H1 / 8; c += THREADS) {
          int r = c >> 4, cc = (c & 15) << 3;
          st8(d2 + r * SW2_STRIDE + cc, ld8(s2 + r * H1 + cc));
        }
        const h16* s3 = w3t + (size_t)k * ADIM * H2;
        h16* d3 = smem + W3B_BASE + (k & 1) * W3BUF;
        for (int c = tid; c < ADIM * H2 / 8; c += THREADS) {
          int r = c >> 3, cc = (c & 7) << 3;
          st8(d3 + r * SW3_STRIDE + cc, ld8(s3 + r * H2 + cc));
        }
      }
    }
#endif
    __syncthreads();  // slot t data visible to all waves

    // ---- GEMM1 half: [16 x 512] x [512 x 64] into h1[half*4 .. half*4+3] ----
    const h16* w1buf = smem + W1B_BASE + half * W1HALF;
    if (half == 0) {
      #pragma unroll
      for (int i = 0; i < 8; ++i) h1[i] = (v8f){};
    }
    for (int ks = 0; ks < 16; ++ks) {
      v16h a = ldA(sA, wave * 16 + ln16, SA_STRIDE, ks * 32, sel);
      #pragma unroll
      for (int nt = 0; nt < 4; ++nt) {
        v16h b = ldB(w1buf, nt * 16 + ln16, SW1_STRIDE, ks * 32, sel);
        h1[half * 4 + nt] = WMMA_F16(a, b, h1[half * 4 + nt]);
      }
    }
    if (half == 0) continue;  // second half of W1 still needed for this expert

    // ---- bias + SELU + LayerNorm (row M = v + 8*sel across 16-lane group) ----
    const h16* w2buf = smem + W2B_BASE + (k & 1) * W2BUF;
    const h16* w3buf = smem + W3B_BASE + (k & 1) * W3BUF;
    float s[8] = {}, q[8] = {};
    #pragma unroll
    for (int nt = 0; nt < 8; ++nt) {
      float bb = b1[k * H1 + nt * 16 + ln16];
      #pragma unroll
      for (int v = 0; v < 8; ++v) {
        float x = selu_f(h1[nt][v] + bb);
        h1[nt][v] = x;
        s[v] += x; q[v] += x * x;
      }
    }
    #pragma unroll
    for (int v = 0; v < 8; ++v) {
      #pragma unroll
      for (int m = 8; m >= 1; m >>= 1) {
        s[v] += __shfl_xor(s[v], m, 32);
        q[v] += __shfl_xor(q[v], m, 32);
      }
    }
    #pragma unroll
    for (int nt = 0; nt < 8; ++nt) {
      int col = nt * 16 + ln16;
      float ga = lng[k * H1 + col], be = lnb[k * H1 + col];
      #pragma unroll
      for (int v = 0; v < 8; ++v) {
        float mean = s[v] * (1.0f / H1);
        float var  = q[v] * (1.0f / H1) - mean * mean;
        float inv  = rsqrtf(var + 1e-5f);
        float y = (h1[nt][v] - mean) * inv * ga + be;
        sH[(v + 8 * sel) * SH_STRIDE + col] = (h16)y;  // D-layout -> A-layout
      }
    }

    // ---- GEMM2: [16 x 128] x [128 x 64] ----
    v8f h2[4] = {};
    for (int ks = 0; ks < 4; ++ks) {
      v16h a = ldA(sH, ln16, SH_STRIDE, ks * 32, sel);
      #pragma unroll
      for (int nt = 0; nt < 4; ++nt) {
        v16h b = ldB(w2buf, nt * 16 + ln16, SW2_STRIDE, ks * 32, sel);
        h2[nt] = WMMA_F16(a, b, h2[nt]);
      }
    }
    #pragma unroll
    for (int nt = 0; nt < 4; ++nt) {
      float bb = b2[k * H2 + nt * 16 + ln16];
      #pragma unroll
      for (int v = 0; v < 8; ++v) {
        float x = selu_f(h2[nt][v] + bb);
        sH[(v + 8 * sel) * SH_STRIDE + nt * 16 + ln16] = (h16)x;  // reuse scratch
      }
    }

    // ---- GEMM3: [16 x 64] x [64 x 64] ----
    v8f o[4] = {};
    for (int ks = 0; ks < 2; ++ks) {
      v16h a = ldA(sH, ln16, SH_STRIDE, ks * 32, sel);
      #pragma unroll
      for (int nt = 0; nt < 4; ++nt) {
        v16h b = ldB(w3buf, nt * 16 + ln16, SW3_STRIDE, ks * 32, sel);
        o[nt] = WMMA_F16(a, b, o[nt]);
      }
    }
    // ---- bias + tanh + gated accumulation (row m's gate-k at lane (G*16)|k) ----
    #pragma unroll
    for (int nt = 0; nt < 4; ++nt) {
      float bb = b3[k * ADIM + nt * 16 + ln16];
      #pragma unroll
      for (int v = 0; v < 8; ++v) {
        float x = o[nt][v] + bb;
        x = fminf(fmaxf(x, -15.0f), 15.0f);
        float e = __expf(2.0f * x);
        float tt = (e - 1.0f) / (e + 1.0f);
        float gv = __shfl(g[v], (lane & 16) | k, 32);
        accOut[nt][v] += gv * tt;
      }
    }
  }

  // ---- write gated output tile ----
  #pragma unroll
  for (int nt = 0; nt < 4; ++nt) {
    #pragma unroll
    for (int v = 0; v < 8; ++v) {
      out[(rowBase + wave * 16 + v + 8 * sel) * ADIM + nt * 16 + ln16] = accOut[nt][v];
    }
  }
}

// ---------------------------------------------------------------------------
extern "C" void kernel_launch(void* const* d_in, const int* in_sizes, int n_in,
                              void* d_out, int out_size, void* d_ws, size_t ws_size,
                              hipStream_t stream) {
  (void)in_sizes; (void)n_in; (void)out_size; (void)ws_size;
  const float* state = (const float*)d_in[0];
  const float* W1    = (const float*)d_in[1];
  const float* b1    = (const float*)d_in[2];
  const float* lng   = (const float*)d_in[3];
  const float* lnb   = (const float*)d_in[4];
  const float* W2    = (const float*)d_in[5];
  const float* b2    = (const float*)d_in[6];
  const float* W3    = (const float*)d_in[7];
  const float* b3    = (const float*)d_in[8];
  const float* gW    = (const float*)d_in[9];
  const float* gb    = (const float*)d_in[10];
  float* out = (float*)d_out;

  char* ws = (char*)d_ws;
  h16* w1t = (h16*)(ws + WS_W1T);
  h16* w2t = (h16*)(ws + WS_W2T);
  h16* w3t = (h16*)(ws + WS_W3T);
  h16* gwt = (h16*)(ws + WS_GWT);
  h16* stf = (h16*)(ws + WS_STATE);

  hipLaunchKernelGGL(pack_inputs_kernel, dim3(1024), dim3(256), 0, stream,
                     state, W1, W2, W3, gW, w1t, w2t, w3t, gwt, stf);

  hipFuncSetAttribute((const void*)moe_fused_kernel,
                      hipFuncAttributeMaxDynamicSharedMemorySize, SMEM_BYTES);
  hipLaunchKernelGGL(moe_fused_kernel, dim3(NBLOCKS), dim3(THREADS), SMEM_BYTES, stream,
                     stf, b1, lng, lnb, b2, b3, gb, w1t, w2t, w3t, gwt, out);
}
